// MultiheadAttention_60928406061457
// MI455X (gfx1250) — matrix operand order repaired and checked by reading, not compiled, not run on hardware
//
#include <hip/hip_runtime.h>

#define D_MODEL 768
#define HEADS   12
#define DK      64
#define BATCH   4
#define LSEQ    2048
#define SCALING 0.35355339059327379f  // 64^-0.25

typedef __attribute__((ext_vector_type(16))) __bf16 bf16x16;
typedef __attribute__((ext_vector_type(2)))  __bf16 bf16x2;
typedef __attribute__((ext_vector_type(8)))  float  f32x8;

union ABFrag {
  unsigned int u[8];
  bf16x16 v;
};

__device__ __forceinline__ unsigned short f32_to_bf16(float f) {
  union { float f; unsigned int u; } c; c.f = f;
  unsigned int r = c.u + 0x7fffu + ((c.u >> 16) & 1u);  // round-to-nearest-even
  return (unsigned short)(r >> 16);
}

// Packed f32x2 -> bf16x2 (v_cvt_pk_bf16_f32 when available; 1 VALU op / 2 elems)
__device__ __forceinline__ unsigned int pack_bf16(float lo, float hi) {
#if __has_builtin(__builtin_amdgcn_cvt_pk_bf16_f32)
  bf16x2 p = __builtin_amdgcn_cvt_pk_bf16_f32(lo, hi);
  return __builtin_bit_cast(unsigned int, p);
#else
  return (unsigned int)f32_to_bf16(lo) | ((unsigned int)f32_to_bf16(hi) << 16);
#endif
}

// ---- CDNA5 async copy: global -> LDS, 16B per lane, tracked by ASYNCcnt ----
// Generic pointers to LDS keep the LDS byte offset in the low 32 bits
// (aperture check discards the upper bits, ISA 10.2), so truncating the
// flat address yields the VDST LDS-address operand.
__device__ __forceinline__ void async_b128(unsigned short* lds_dst,
                                           const unsigned short* src) {
  unsigned int loff = (unsigned int)(size_t)lds_dst;
  unsigned long long ga = (unsigned long long)(size_t)src;
  asm volatile("global_load_async_to_lds_b128 %0, %1, off"
               :: "v"(loff), "v"(ga) : "memory");
}

__device__ __forceinline__ void wait_async_le4() {
#if __has_builtin(__builtin_amdgcn_s_wait_asynccnt)
  __builtin_amdgcn_s_wait_asynccnt(4);
#else
  asm volatile("s_wait_asynccnt 0x4" ::: "memory");
#endif
}
__device__ __forceinline__ void wait_async_le0() {
#if __has_builtin(__builtin_amdgcn_s_wait_asynccnt)
  __builtin_amdgcn_s_wait_asynccnt(0);
#else
  asm volatile("s_wait_asynccnt 0x0" ::: "memory");
#endif
}

// A fragment (16x32 bf16, MxK), per ISA 7.12.2.
__device__ __forceinline__ void load_frag_a(const unsigned short* base, int stride,
                                            unsigned int* regs) {
  const int lane = threadIdx.x & 31;
  const int hl   = lane >> 4;
  const int r    = lane & 15;
  const unsigned short* p = base + r * stride + hl * 8;
#pragma unroll
  for (int v = 0; v < 8; ++v) {
    const int k = ((v & 3) << 1) + ((v >> 2) << 4);
    regs[v] = *(const unsigned int*)(p + k);
  }
}

// B fragment (32x16 bf16, KxN) from an [N][K]-major tile.
__device__ __forceinline__ void load_frag_b(const unsigned short* base, int stride,
                                            unsigned int* regs) {
  const int lane = threadIdx.x & 31;
  const int hl   = lane >> 4;
  const int n    = lane & 15;
  const unsigned short* p = base + n * stride + hl * 8;
#pragma unroll
  for (int v = 0; v < 8; ++v) regs[v] = *(const unsigned int*)(p + ((v & 3) << 1) + ((v >> 2) << 4));
}

__device__ __forceinline__ f32x8 wmma_bf16(const ABFrag& a, const ABFrag& b, f32x8 c) {
  return __builtin_amdgcn_wmma_f32_16x16x32_bf16(false, a.v, false, b.v,
                                                 (short)0, c, false, false);
}

// ---------------------------------------------------------------------------
// Projection GEMM: out (bf16) = scale * (A[8192x768] @ W[768x768])
// DMAJOR=0 -> [B,H,L,DK] (Q,K); DMAJOR=1 -> [B,H,DK,L] (V, async-friendly V^T)
// ---------------------------------------------------------------------------
template <int DMAJOR>
__global__ __launch_bounds__(256) void qkv_proj_kernel(
    const float* __restrict__ A, const float* __restrict__ W,
    unsigned short* __restrict__ outQ, float scale)
{
  __shared__ __align__(16) unsigned short Al[128 * 40];
  __shared__ __align__(16) unsigned short Bt[128 * 40];  // [n][k]
  const int tid  = threadIdx.x;
  const int m0   = blockIdx.x * 128;
  const int n0   = blockIdx.y * 128;
  const int wave = tid >> 5;
  const int wr = wave >> 2, wc = wave & 3;
  const int lane = tid & 31, hl = lane >> 4, ln = lane & 15;

  f32x8 zero = {};
  f32x8 c[4][2];
#pragma unroll
  for (int i = 0; i < 4; ++i)
#pragma unroll
    for (int j = 0; j < 2; ++j) c[i][j] = zero;

  for (int k0 = 0; k0 < D_MODEL; k0 += 32) {
    {  // stage A tile 128x32, f32 -> bf16 (packed converts)
      const int rA = tid >> 3;
      const int kA = (tid & 7) * 4;
#pragma unroll
      for (int i = 0; i < 4; ++i) {
        const int r = rA + 32 * i;
        const float4 d = *(const float4*)(A + (size_t)(m0 + r) * D_MODEL + k0 + kA);
        *(unsigned int*)&Al[r * 40 + kA]     = pack_bf16(d.x, d.y);
        *(unsigned int*)&Al[r * 40 + kA + 2] = pack_bf16(d.z, d.w);
      }
    }
    {  // stage B tile 32x128 transposed -> Bt[n][k]
      const int kB = tid >> 5;
      const int cB = (tid & 31) * 4;
#pragma unroll
      for (int i = 0; i < 4; ++i) {
        const int k = kB + 8 * i;
        const float4 d = *(const float4*)(W + (size_t)(k0 + k) * D_MODEL + n0 + cB);
        Bt[(cB + 0) * 40 + k] = f32_to_bf16(d.x);
        Bt[(cB + 1) * 40 + k] = f32_to_bf16(d.y);
        Bt[(cB + 2) * 40 + k] = f32_to_bf16(d.z);
        Bt[(cB + 3) * 40 + k] = f32_to_bf16(d.w);
      }
    }
    __syncthreads();
    ABFrag af[4], bf[2];
#pragma unroll
    for (int mt = 0; mt < 4; ++mt)
      load_frag_a(&Al[(wr * 64 + mt * 16) * 40], 40, af[mt].u);
#pragma unroll
    for (int nt = 0; nt < 2; ++nt)
      load_frag_b(&Bt[(wc * 32 + nt * 16) * 40], 40, bf[nt].u);
#pragma unroll
    for (int mt = 0; mt < 4; ++mt)
#pragma unroll
      for (int nt = 0; nt < 2; ++nt)
        c[mt][nt] = wmma_bf16(af[mt], bf[nt], c[mt][nt]);
    __syncthreads();
  }

#pragma unroll
  for (int mt = 0; mt < 4; ++mt)
#pragma unroll
    for (int nt = 0; nt < 2; ++nt)
#pragma unroll
      for (int r = 0; r < 8; ++r) {
        const int M = m0 + wr * 64 + mt * 16 + r + 8 * hl;
        const int N = n0 + wc * 32 + nt * 16 + ln;
        const int b = M >> 11, l = M & (LSEQ - 1);
        const int h = N >> 6,  d = N & 63;
        const size_t idx = DMAJOR
            ? ((size_t)(b * HEADS + h) * DK + d) * LSEQ + l
            : ((size_t)(b * HEADS + h) * LSEQ + l) * DK + d;
        outQ[idx] = f32_to_bf16(c[mt][nt][r] * scale);
      }
}

// ---------------------------------------------------------------------------
// Fused flash attention with double-buffered async K/V staging.
// Q,K workspace: [B,H,L,DK]; V workspace: [B,H,DK,L] (pre-transposed).
// ---------------------------------------------------------------------------
__global__ __launch_bounds__(256) void attention_kernel(
    const unsigned short* __restrict__ Qb, const unsigned short* __restrict__ Kb,
    const unsigned short* __restrict__ Vb, const float* __restrict__ posbias,
    unsigned short* __restrict__ ctx)
{
  __shared__ __align__(16) unsigned short Kl[2][64 * 72];   // [j][d]
  __shared__ __align__(16) unsigned short Vt[2][64 * 72];   // [d][j]
  __shared__ __align__(16) unsigned short Pl[8 * 16 * 72];  // per-wave P tile

  const int tid  = threadIdx.x;
  const int wave = tid >> 5;
  const int lane = tid & 31, hl = lane >> 4, ln = lane & 15;
  const int bh = blockIdx.y;
  const int b = bh / HEADS, h = bh % HEADS;
  const int q0 = blockIdx.x * 128;

  const unsigned short* Qbh = Qb + (size_t)bh * LSEQ * DK;
  const unsigned short* Kbh = Kb + (size_t)bh * LSEQ * DK;
  const unsigned short* Vbh = Vb + (size_t)bh * DK * LSEQ;  // d-major
  const float* pb = posbias + ((size_t)h * LSEQ + q0) * LSEQ;

  // async stage of one 64-key tile: K rows (j-major) + V^T rows (d-major)
  auto stage_tile = [&](int buf, int j0n) {
    const int rr0 = tid >> 3;
    const int cq  = (tid & 7) * 8;
#pragma unroll
    for (int i = 0; i < 2; ++i) {
      const int rr = rr0 + 32 * i;
      async_b128(&Kl[buf][rr * 72 + cq], Kbh + (size_t)(j0n + rr) * DK + cq);
      async_b128(&Vt[buf][rr * 72 + cq], Vbh + (size_t)rr * LSEQ + j0n + cq);
    }
  };

  // Q fragments for this wave's 16 rows, kept in registers for all 32 tiles
  ABFrag qa[2];
  {
    const unsigned short* qp = Qbh + (size_t)(q0 + wave * 16) * DK;
    load_frag_a(qp,      DK, qa[0].u);
    load_frag_a(qp + 32, DK, qa[1].u);
  }

  f32x8 zero = {};
  f32x8 o[4];
#pragma unroll
  for (int i = 0; i < 4; ++i) o[i] = zero;
  float m_acc[8], l_acc[8];
#pragma unroll
  for (int r = 0; r < 8; ++r) { m_acc[r] = -1e30f; l_acc[r] = 0.f; }

  unsigned short* Pw = &Pl[wave * 16 * 72];

  constexpr int NT = LSEQ / 64;  // 32 tiles
  stage_tile(0, 0);
  stage_tile(1, 64);

  for (int it = 0; it < NT; ++it) {
    const int j0  = it * 64;
    const int cur = it & 1;
    if (it + 1 < NT) wait_async_le4(); else wait_async_le0();
    __syncthreads();

    // stream-ahead the next tile's posbias rows (32 lanes x 128B lines)
    if (it + 1 < NT) {
      const int prow = wave * 16 + (lane & 15);
      __builtin_prefetch(&pb[(size_t)prow * LSEQ + j0 + 64 + ((lane >> 4) << 5)], 0, 0);
    }

    const unsigned short* Kc = Kl[cur];
    const unsigned short* Vc = Vt[cur];

    // S = Q K^T  (16 x 64)
    f32x8 s[4];
#pragma unroll
    for (int nt = 0; nt < 4; ++nt) {
      ABFrag kf;
      load_frag_b(&Kc[(nt * 16) * 72], 72, kf.u);        // d = 0..31
      s[nt] = wmma_bf16(qa[0], kf, zero);
      load_frag_b(&Kc[(nt * 16) * 72 + 32], 72, kf.u);   // d = 32..63
      s[nt] = wmma_bf16(qa[1], kf, s[nt]);
    }

    // + posbias, per-row online softmax
    const int qrb = wave * 16 + 8 * hl;
    float rmax[8];
#pragma unroll
    for (int r = 0; r < 8; ++r) rmax[r] = -1e30f;
#pragma unroll
    for (int nt = 0; nt < 4; ++nt)
#pragma unroll
      for (int r = 0; r < 8; ++r) {
        const float v =
            s[nt][r] + pb[(size_t)(qrb + r) * LSEQ + j0 + nt * 16 + ln];
        s[nt][r] = v;
        rmax[r] = fmaxf(rmax[r], v);
      }
#pragma unroll
    for (int r = 0; r < 8; ++r) {
      float v = rmax[r];
      v = fmaxf(v, __shfl_xor(v, 1, 32));
      v = fmaxf(v, __shfl_xor(v, 2, 32));
      v = fmaxf(v, __shfl_xor(v, 4, 32));
      v = fmaxf(v, __shfl_xor(v, 8, 32));
      rmax[r] = v;
    }
    float alpha[8], rsum[8];
#pragma unroll
    for (int r = 0; r < 8; ++r) {
      const float mnew = fmaxf(m_acc[r], rmax[r]);
      alpha[r] = __expf(m_acc[r] - mnew);
      m_acc[r] = mnew;
      rsum[r] = 0.f;
    }
#pragma unroll
    for (int nt = 0; nt < 4; ++nt)
#pragma unroll
      for (int r = 0; r < 8; ++r) {
        const float p = __expf(s[nt][r] - m_acc[r]);
        s[nt][r] = p;
        rsum[r] += p;
      }
#pragma unroll
    for (int r = 0; r < 8; ++r) {
      float v = rsum[r];
      v += __shfl_xor(v, 1, 32);
      v += __shfl_xor(v, 2, 32);
      v += __shfl_xor(v, 4, 32);
      v += __shfl_xor(v, 8, 32);
      l_acc[r] = l_acc[r] * alpha[r] + v;
    }
#pragma unroll
    for (int nt = 0; nt < 4; ++nt)
#pragma unroll
      for (int r = 0; r < 8; ++r) o[nt][r] *= alpha[r];

    // P -> per-wave LDS (bf16) to re-enter A-fragment layout
#pragma unroll
    for (int nt = 0; nt < 4; ++nt)
#pragma unroll
      for (int r = 0; r < 8; ++r)
        Pw[(r + 8 * hl) * 72 + nt * 16 + ln] = f32_to_bf16(s[nt][r]);

    // O += P V
#pragma unroll
    for (int ks = 0; ks < 2; ++ks) {
      ABFrag pa;
      load_frag_a(Pw + ks * 32, 72, pa.u);
#pragma unroll
      for (int nt = 0; nt < 4; ++nt) {
        ABFrag vf;
        load_frag_b(&Vc[(nt * 16) * 72 + ks * 32], 72, vf.u);
        o[nt] = wmma_bf16(pa, vf, o[nt]);
      }
    }
    __syncthreads();  // everyone done with buffer `cur` before refill
    if (it + 2 < NT) stage_tile(cur, j0 + 128);
  }

  // normalize, store context bf16 as [B][L][D_MODEL]
  float inv[8];
#pragma unroll
  for (int r = 0; r < 8; ++r) inv[r] = 1.0f / l_acc[r];
#pragma unroll
  for (int nt = 0; nt < 4; ++nt)
#pragma unroll
    for (int r = 0; r < 8; ++r) {
      const int l = q0 + wave * 16 + r + 8 * hl;
      const int d = nt * 16 + ln;
      ctx[(size_t)(b * LSEQ + l) * D_MODEL + h * DK + d] =
          f32_to_bf16(o[nt][r] * inv[r]);
    }
}

// ---------------------------------------------------------------------------
// Output projection: out (f32) = ctx(bf16) @ Wo + bo   (async A staging)
// ---------------------------------------------------------------------------
__global__ __launch_bounds__(256) void out_proj_kernel(
    const unsigned short* __restrict__ Actx, const float* __restrict__ W,
    const float* __restrict__ bo, float* __restrict__ out)
{
  __shared__ __align__(16) unsigned short Al[128 * 40];
  __shared__ __align__(16) unsigned short Bt[128 * 40];
  const int tid  = threadIdx.x;
  const int m0   = blockIdx.x * 128;
  const int n0   = blockIdx.y * 128;
  const int wave = tid >> 5;
  const int wr = wave >> 2, wc = wave & 3;
  const int lane = tid & 31, hl = lane >> 4, ln = lane & 15;

  f32x8 zero = {};
  f32x8 c[4][2];
#pragma unroll
  for (int i = 0; i < 4; ++i)
#pragma unroll
    for (int j = 0; j < 2; ++j) c[i][j] = zero;

  for (int k0 = 0; k0 < D_MODEL; k0 += 32) {
    {  // A tile already bf16: async global->LDS copies
      const int rA = tid >> 2;
      const int kA = (tid & 3) * 8;
#pragma unroll
      for (int i = 0; i < 2; ++i) {
        const int r = rA + 64 * i;
        async_b128(&Al[r * 40 + kA], Actx + (size_t)(m0 + r) * D_MODEL + k0 + kA);
      }
    }
    {  // Wo f32 -> bf16 transposed
      const int kB = tid >> 5;
      const int cB = (tid & 31) * 4;
#pragma unroll
      for (int i = 0; i < 4; ++i) {
        const int k = kB + 8 * i;
        const float4 d = *(const float4*)(W + (size_t)(k0 + k) * D_MODEL + n0 + cB);
        Bt[(cB + 0) * 40 + k] = f32_to_bf16(d.x);
        Bt[(cB + 1) * 40 + k] = f32_to_bf16(d.y);
        Bt[(cB + 2) * 40 + k] = f32_to_bf16(d.z);
        Bt[(cB + 3) * 40 + k] = f32_to_bf16(d.w);
      }
    }
    wait_async_le0();
    __syncthreads();
    ABFrag af[4], bf[2];
#pragma unroll
    for (int mt = 0; mt < 4; ++mt)
      load_frag_a(&Al[(wr * 64 + mt * 16) * 40], 40, af[mt].u);
#pragma unroll
    for (int nt = 0; nt < 2; ++nt)
      load_frag_b(&Bt[(wc * 32 + nt * 16) * 40], 40, bf[nt].u);
#pragma unroll
    for (int mt = 0; mt < 4; ++mt)
#pragma unroll
      for (int nt = 0; nt < 2; ++nt)
        c[mt][nt] = wmma_bf16(af[mt], bf[nt], c[mt][nt]);
    __syncthreads();
  }

  float bv[2];
#pragma unroll
  for (int nt = 0; nt < 2; ++nt) bv[nt] = bo[n0 + wc * 32 + nt * 16 + ln];
#pragma unroll
  for (int mt = 0; mt < 4; ++mt)
#pragma unroll
    for (int nt = 0; nt < 2; ++nt)
#pragma unroll
      for (int r = 0; r < 8; ++r) {
        const int M = m0 + wr * 64 + mt * 16 + r + 8 * hl;
        const int N = n0 + wc * 32 + nt * 16 + ln;
        out[(size_t)M * D_MODEL + N] = c[mt][nt][r] + bv[nt];
      }
}

extern "C" void kernel_launch(void* const* d_in, const int* in_sizes, int n_in,
                              void* d_out, int out_size, void* d_ws, size_t ws_size,
                              hipStream_t stream) {
  (void)in_sizes; (void)n_in; (void)out_size; (void)ws_size;
  const float* query   = (const float*)d_in[0];
  const float* key_    = (const float*)d_in[1];
  const float* value   = (const float*)d_in[2];
  const float* posbias = (const float*)d_in[3];
  const float* Wq      = (const float*)d_in[4];
  const float* Wk      = (const float*)d_in[5];
  const float* Wv      = (const float*)d_in[6];
  const float* Wo      = (const float*)d_in[7];
  const float* bo      = (const float*)d_in[8];

  // workspace: Q/K [B,H,L,64], V [B,H,64,L], ctx [B,L,768] -- all bf16 (~50 MB)
  const size_t per = (size_t)BATCH * HEADS * LSEQ * DK;
  unsigned short* Qb  = (unsigned short*)d_ws;
  unsigned short* Kb  = Qb + per;
  unsigned short* Vb  = Kb + per;
  unsigned short* ctx = Vb + per;

  dim3 blk(256);
  dim3 gP(64, 6);  // (8192/128, 768/128)
  qkv_proj_kernel<0><<<gP, blk, 0, stream>>>(query, Wq, Qb, SCALING);
  qkv_proj_kernel<0><<<gP, blk, 0, stream>>>(key_,  Wk, Kb, SCALING);
  qkv_proj_kernel<1><<<gP, blk, 0, stream>>>(value, Wv, Vb, 1.0f);
  attention_kernel<<<dim3(16, BATCH * HEADS), blk, 0, stream>>>(Qb, Kb, Vb, posbias, ctx);
  out_proj_kernel<<<gP, blk, 0, stream>>>(ctx, Wo, bo, (float*)d_out);
}
